// AdaptiveTSSA_Enhanced_11802570129747
// MI455X (gfx1250) — compile-verified, matrix operand order checked by reading, not run on hardware
//
#include <hip/hip_runtime.h>
#include <math.h>

// ---------------- problem constants ----------------
#define BDIM 4
#define CDIM 256
#define NPIX 1024      // 32*32
#define LSEQ 3072      // 3 scales * NPIX
#define NHEADS 4
#define HD 64

typedef __attribute__((ext_vector_type(16))) _Float16 v16h;
typedef __attribute__((ext_vector_type(8)))  _Float16 v8h;
typedef __attribute__((ext_vector_type(8)))  float    v8f;
typedef __attribute__((ext_vector_type(4)))  float    v4f;

union FragA { v16h v; _Float16 h[16]; unsigned u[8]; };
union FragC { v8f  v; float f[8]; };

// =====================================================================
// WMMA GEMM:  Y[b] (MxN) = W (MxK) @ X[b] (KxN)   (+bias per M, +addsrc)
// X rows < ksplit come from X0, rows >= ksplit from X1 (concat support).
// Block tile 32(M) x 128(N), K-step 32; 8 waves, each wave 16x32 output
// (2 accumulators -> 2 v_wmma per K-step). Staging is float4-vectorized,
// branchless and register-prefetched (software pipelined).
// All of M%32==0, N%128==0, K%32==0 hold for every call site.
// =====================================================================
__global__ __launch_bounds__(256)
void gemm_wmma_kernel(const float* __restrict__ Wm,
                      const float* __restrict__ X0,
                      const float* __restrict__ X1, int ksplit,
                      float* __restrict__ Y,
                      const float* __restrict__ bias,
                      const float* __restrict__ addsrc,
                      int M, int N, int K,
                      long sx0, long sx1, long sy, long sadd)
{
    __shared__ _Float16 lds_a[32 * 32];          // [m][k]
    __shared__ _Float16 lds_b[16 * 128 * 2];     // [kpair][n][parity]
    const int b = blockIdx.z;
    const int rowBase = blockIdx.y * 32;
    const int colBase = blockIdx.x * 128;
    const int tid = threadIdx.x;
    const int wv = tid >> 5, lane = tid & 31;
    const int wm = wv >> 2, wn = wv & 3;
    const int kg = lane >> 4, lx = lane & 15;
    const float* x0b = X0 + (long)b * sx0;
    const float* x1b = X1 ? (X1 + (long)b * sx1) : x0b;

    // staging assignments
    const int a_m  = tid >> 3;           // 0..31
    const int a_k4 = (tid & 7) * 4;      // k quad within 32
    const int bkp0 = tid >> 5;           // k-pair rows: task0 -> 0..7
    const int bkp1 = bkp0 + 8;           //              task1 -> 8..15
    const int bn0  = (tid & 31) * 4;     // n quad within 128

    FragC acc0, acc1;
    #pragma unroll
    for (int j = 0; j < 8; ++j) { acc0.f[j] = 0.f; acc1.f[j] = 0.f; }

    v4f ra;          // W prefetch
    v4f rb[2][2];    // X prefetch: 2 tasks x 2 k-parity rows

    // ---- prologue: prefetch k0 = 0 ----
    {
        ra = *(const v4f*)&Wm[(long)(rowBase + a_m) * K + a_k4];
        #pragma unroll
        for (int t = 0; t < 2; ++t) {
            int kp = (t == 0) ? bkp0 : bkp1;
            int kr0 = 2 * kp;
            const float* r0 = (kr0     < ksplit) ? x0b + (long)kr0 * N
                                                 : x1b + (long)(kr0 - ksplit) * N;
            const float* r1 = (kr0 + 1 < ksplit) ? x0b + (long)(kr0 + 1) * N
                                                 : x1b + (long)(kr0 + 1 - ksplit) * N;
            rb[t][0] = *(const v4f*)&r0[colBase + bn0];
            rb[t][1] = *(const v4f*)&r1[colBase + bn0];
        }
    }

    for (int k0 = 0; k0 < K; k0 += 32) {
        // ---- commit staged registers to LDS (f32 -> f16) ----
        {
            union { _Float16 h[4]; unsigned long long q; } pa;
            #pragma unroll
            for (int i = 0; i < 4; ++i) pa.h[i] = (_Float16)ra[i];
            *(unsigned long long*)&lds_a[a_m * 32 + a_k4] = pa.q;
            #pragma unroll
            for (int t = 0; t < 2; ++t) {
                int kp = (t == 0) ? bkp0 : bkp1;
                v8h hh;
                #pragma unroll
                for (int i = 0; i < 4; ++i) {
                    hh[2 * i]     = (_Float16)rb[t][0][i];
                    hh[2 * i + 1] = (_Float16)rb[t][1][i];
                }
                *(v8h*)&lds_b[(kp * 128 + bn0) * 2] = hh;
            }
        }
        __syncthreads();

        // ---- prefetch next K-step while this one computes ----
        if (k0 + 32 < K) {
            int kn = k0 + 32;
            ra = *(const v4f*)&Wm[(long)(rowBase + a_m) * K + kn + a_k4];
            #pragma unroll
            for (int t = 0; t < 2; ++t) {
                int kp = (t == 0) ? bkp0 : bkp1;
                int kr0 = kn + 2 * kp;
                const float* r0 = (kr0     < ksplit) ? x0b + (long)kr0 * N
                                                     : x1b + (long)(kr0 - ksplit) * N;
                const float* r1 = (kr0 + 1 < ksplit) ? x0b + (long)(kr0 + 1) * N
                                                     : x1b + (long)(kr0 + 1 - ksplit) * N;
                rb[t][0] = *(const v4f*)&r0[colBase + bn0];
                rb[t][1] = *(const v4f*)&r1[colBase + bn0];
            }
        }

        // ---- fragments + WMMA ----
        FragA fa, fb0, fb1;
        {
            const unsigned* pa = (const unsigned*)&lds_a[(lx + wm * 16) * 32];
            const unsigned* pb = (const unsigned*)lds_b;
            const int n0 = wn * 32 + lx;
            #pragma unroll
            for (int j = 0; j < 8; ++j) {
                int kpj = (j & 3) + 4 * kg + 8 * (j >> 2);   // k-pair index
                fa.u[j]  = pa[kpj];
                fb0.u[j] = pb[kpj * 128 + n0];
                fb1.u[j] = pb[kpj * 128 + n0 + 16];
            }
        }
        acc0.v = __builtin_amdgcn_wmma_f32_16x16x32_f16(false, fa.v, false, fb0.v,
                                                        (short)0, acc0.v, false, false);
        acc1.v = __builtin_amdgcn_wmma_f32_16x16x32_f16(false, fa.v, false, fb1.v,
                                                        (short)0, acc1.v, false, false);
        __syncthreads();
    }

    // ---- epilogue (ISA C-layout: lane holds rows j+8*kg, col lx) ----
    #pragma unroll
    for (int j = 0; j < 8; ++j) {
        int gm  = rowBase + wm * 16 + j + 8 * kg;
        int gn0 = colBase + wn * 32 + lx;
        float v0 = acc0.f[j], v1 = acc1.f[j];
        if (bias) { float bb = bias[gm]; v0 += bb; v1 += bb; }
        if (addsrc) {
            v0 += addsrc[(long)b * sadd + (long)gm * N + gn0];
            v1 += addsrc[(long)b * sadd + (long)gm * N + gn0 + 16];
        }
        Y[(long)b * sy + (long)gm * N + gn0]      = v0;
        Y[(long)b * sy + (long)gm * N + gn0 + 16] = v1;
    }
}

// =====================================================================
// Flash attention (4 heads, hd=64) over QKV stored channel-major:
// QKV[b] is (768 x L); q rows [0,256), k rows [256,512), v rows [512,768).
// Each wave: 16 queries; block of 8 waves: 128 queries.
// =====================================================================
__global__ __launch_bounds__(256)
void flash_attn_kernel(const float* __restrict__ QKV, float* __restrict__ O,
                       int L, float scale)
{
    __shared__ _Float16 lds_p[8 * 16 * 32];     // per-wave P tiles
    const int b = blockIdx.z, h = blockIdx.y;
    const int wv = threadIdx.x >> 5, lane = threadIdx.x & 31;
    const int kg = lane >> 4, lx = lane & 15;
    const int qbase = blockIdx.x * 128 + wv * 16;
    const long bb = (long)b * 768 * L;
    const float* Qp = QKV + bb + (long)(h * HD) * L;
    const float* Kp = QKV + bb + (long)(CDIM + h * HD) * L;
    const float* Vp = QKV + bb + (long)(2 * CDIM + h * HD) * L;

    FragA q0, q1;                                // Q tile (16 x 64), scale folded in
    {
        int m = qbase + lx;
        #pragma unroll
        for (int i = 0; i < 16; ++i) {
            int d = (i & 7) + 8 * kg + 16 * (i >> 3);
            q0.h[i] = (_Float16)(Qp[(long)d * L + m] * scale);
            q1.h[i] = (_Float16)(Qp[(long)(d + 32) * L + m] * scale);
        }
    }
    FragC o0, o1, o2, o3;
    float mrow[8], lrow[8];
    #pragma unroll
    for (int j = 0; j < 8; ++j) {
        o0.f[j] = o1.f[j] = o2.f[j] = o3.f[j] = 0.f;
        mrow[j] = -1e30f; lrow[j] = 0.f;
    }
    _Float16* myp = &lds_p[wv * 512];

    for (int kb = 0; kb < L; kb += 32) {
        FragC s[2];
        #pragma unroll
        for (int t = 0; t < 2; ++t) {            // two 16-key tiles
            FragA bk;
            int key = kb + t * 16 + lx;
            #pragma unroll
            for (int i = 0; i < 16; ++i) {
                int d = (i & 7) + 8 * kg + 16 * (i >> 3);
                bk.h[i] = (_Float16)Kp[(long)d * L + key];
            }
            FragC z;
            #pragma unroll
            for (int j = 0; j < 8; ++j) z.f[j] = 0.f;
            v8f tmp = __builtin_amdgcn_wmma_f32_16x16x32_f16(false, q0.v, false, bk.v,
                                                             (short)0, z.v, false, false);
            #pragma unroll
            for (int i = 0; i < 16; ++i) {
                int d = 32 + (i & 7) + 8 * kg + 16 * (i >> 3);
                bk.h[i] = (_Float16)Kp[(long)d * L + key];
            }
            s[t].v = __builtin_amdgcn_wmma_f32_16x16x32_f16(false, q1.v, false, bk.v,
                                                            (short)0, tmp, false, false);
        }
        // online softmax row stats (reduce over 16-lane groups)
        #pragma unroll
        for (int j = 0; j < 8; ++j) {
            float vmax = fmaxf(s[0].f[j], s[1].f[j]);
            #pragma unroll
            for (int off = 8; off >= 1; off >>= 1)
                vmax = fmaxf(vmax, __shfl_xor(vmax, off, 32));
            float mnew = fmaxf(mrow[j], vmax);
            float al = __expf(mrow[j] - mnew);
            float p0 = __expf(s[0].f[j] - mnew);
            float p1 = __expf(s[1].f[j] - mnew);
            float ps = p0 + p1;
            #pragma unroll
            for (int off = 8; off >= 1; off >>= 1)
                ps += __shfl_xor(ps, off, 32);
            lrow[j] = lrow[j] * al + ps;
            mrow[j] = mnew;
            o0.f[j] *= al; o1.f[j] *= al; o2.f[j] *= al; o3.f[j] *= al;
            int row = j + 8 * kg;
            myp[row * 32 + lx]      = (_Float16)p0;
            myp[row * 32 + 16 + lx] = (_Float16)p1;
        }
        // P (C-layout) -> A-fragment via wave-private LDS
        FragA fp;
        {
            const unsigned* pp = (const unsigned*)&myp[lx * 32];
            #pragma unroll
            for (int j = 0; j < 8; ++j) fp.u[j] = pp[(j & 3) + 4 * kg + 8 * (j >> 2)];
        }
        // O += P (16x32) @ V (32x64); per-lane k-runs contiguous -> float4 loads
        #pragma unroll
        for (int t = 0; t < 4; ++t) {
            FragA bv;
            const float* vr = Vp + (long)(t * 16 + lx) * L + kb;
            v4f va = *(const v4f*)&vr[8 * kg];
            v4f vb_ = *(const v4f*)&vr[8 * kg + 4];
            v4f vc = *(const v4f*)&vr[16 + 8 * kg];
            v4f vd = *(const v4f*)&vr[16 + 8 * kg + 4];
            #pragma unroll
            for (int i = 0; i < 4; ++i) {
                bv.h[i]      = (_Float16)va[i];
                bv.h[4 + i]  = (_Float16)vb_[i];
                bv.h[8 + i]  = (_Float16)vc[i];
                bv.h[12 + i] = (_Float16)vd[i];
            }
            v8f* op = (t == 0) ? &o0.v : (t == 1) ? &o1.v : (t == 2) ? &o2.v : &o3.v;
            *op = __builtin_amdgcn_wmma_f32_16x16x32_f16(false, fp.v, false, bv.v,
                                                         (short)0, *op, false, false);
        }
    }
    float* Ob = O + (long)b * CDIM * L + (long)(h * HD) * L;
    #pragma unroll
    for (int j = 0; j < 8; ++j) {
        float inv = 1.0f / lrow[j];
        int q = qbase + j + 8 * kg;
        Ob[(long)(lx) * L + q]      = o0.f[j] * inv;
        Ob[(long)(16 + lx) * L + q] = o1.f[j] * inv;
        Ob[(long)(32 + lx) * L + q] = o2.f[j] * inv;
        Ob[(long)(48 + lx) * L + q] = o3.f[j] * inv;
    }
}

// ---------------- elementwise / reduction helpers ----------------
__global__ void mean_hw_kernel(const float* __restrict__ x, float* __restrict__ out) {
    __shared__ float sm[256];
    int bc = blockIdx.x;
    const float* p = x + (long)bc * NPIX;
    float s = 0.f;
    for (int i = threadIdx.x; i < NPIX; i += 256) s += p[i];
    sm[threadIdx.x] = s; __syncthreads();
    for (int st = 128; st > 0; st >>= 1) {
        if (threadIdx.x < st) sm[threadIdx.x] += sm[threadIdx.x + st];
        __syncthreads();
    }
    if (threadIdx.x == 0) out[bc] = sm[0] * (1.0f / NPIX);
}

__global__ void gemv_kernel(const float* __restrict__ Wm, const float* __restrict__ bias,
                            const float* __restrict__ vin, float* __restrict__ vout,
                            int O, int K, int act) {
    int b = blockIdx.x, o = threadIdx.x;
    if (o >= O) return;
    const float* vb = vin + (long)b * K;
    float s = bias ? bias[o] : 0.f;
    for (int k = 0; k < K; ++k) s += Wm[(long)o * K + k] * vb[k];
    if (act == 1) s = fmaxf(s, 0.f);
    else if (act == 2) s = 1.f / (1.f + __expf(-s));
    vout[(long)b * O + o] = s;
}

__global__ void softmax3_kernel(float* __restrict__ v) {
    int b = threadIdx.x; if (b >= BDIM) return;
    float a0 = v[b*3], a1 = v[b*3+1], a2 = v[b*3+2];
    float m = fmaxf(a0, fmaxf(a1, a2));
    float e0 = __expf(a0-m), e1 = __expf(a1-m), e2 = __expf(a2-m);
    float inv = 1.f / (e0 + e1 + e2);
    v[b*3] = e0*inv; v[b*3+1] = e1*inv; v[b*3+2] = e2*inv;
}

__global__ void dyt_apply_kernel(const float* __restrict__ x, const float* __restrict__ alphas,
                                 const float* __restrict__ imp, const float* __restrict__ wt,
                                 const float* __restrict__ bs, float* __restrict__ y) {
    long idx = (long)blockIdx.x * 256 + threadIdx.x;
    long bc = idx >> 10;
    int c = (int)(bc & 255), b = (int)(bc >> 8);
    float v = x[idx], s = 0.f;
    #pragma unroll
    for (int i = 0; i < 3; ++i) s += tanhf(alphas[i] * v) * imp[b * 3 + i];
    y[idx] = s * wt[c] + bs[c];
}

__global__ void avgpool_kernel(const float* __restrict__ x, float* __restrict__ y, int s) {
    int hs = 32 / s, ws_ = 32 / s;
    long idx = (long)blockIdx.x * 256 + threadIdx.x;
    long tot = (long)BDIM * CDIM * hs * ws_;
    if (idx >= tot) return;
    int j = (int)(idx % ws_); long r = idx / ws_;
    int i = (int)(r % hs); long bc = r / hs;
    const float* p = x + bc * NPIX;
    float acc = 0.f;
    for (int di = 0; di < s; ++di)
        for (int dj = 0; dj < s; ++dj)
            acc += p[(i * s + di) * 32 + (j * s + dj)];
    y[idx] = acc / (float)(s * s);
}

__global__ void upsample_kernel(const float* __restrict__ x, float* __restrict__ y, int s) {
    int hs = 32 / s, ws_ = 32 / s;
    long idx = (long)blockIdx.x * 256 + threadIdx.x;
    int j = (int)(idx & 31), i = (int)((idx >> 5) & 31);
    long bc = idx >> 10;
    const float* p = x + bc * hs * ws_;
    float fy = (i + 0.5f) / s - 0.5f, fx = (j + 0.5f) / s - 0.5f;
    int y0 = (int)floorf(fy), x0 = (int)floorf(fx);
    float wy = fy - y0, wx = fx - x0;
    int y1 = min(max(y0 + 1, 0), hs - 1), x1 = min(max(x0 + 1, 0), ws_ - 1);
    y0 = min(max(y0, 0), hs - 1); x0 = min(max(x0, 0), ws_ - 1);
    float v00 = p[y0*ws_+x0], v01 = p[y0*ws_+x1], v10 = p[y1*ws_+x0], v11 = p[y1*ws_+x1];
    y[idx] = v00*(1-wy)*(1-wx) + v01*(1-wy)*wx + v10*wy*(1-wx) + v11*wy*wx;
}

__global__ void tssa_pi_kernel(const float* __restrict__ qkv, const float* __restrict__ temps,
                               int si, float* __restrict__ Pi) {
    __shared__ float logit[NPIX];
    __shared__ float red[256];
    int b = blockIdx.x >> 2, h = blockIdx.x & 3;
    const float* qb = qkv + (long)b * 768 * NPIX + (long)(h * HD) * NPIX;
    float tval = temps[si * NHEADS + h];
    for (int n = threadIdx.x; n < NPIX; n += 256) {
        float s = 0.f;
        for (int d = 0; d < HD; ++d) { float q = qb[(long)d * NPIX + n]; s += q * q; }
        float nrm = fmaxf(sqrtf(s), 1e-12f);
        logit[n] = (s / (nrm * nrm)) * tval;
    }
    __syncthreads();
    float m = -1e30f;
    for (int n = threadIdx.x; n < NPIX; n += 256) m = fmaxf(m, logit[n]);
    red[threadIdx.x] = m; __syncthreads();
    for (int st = 128; st > 0; st >>= 1) {
        if (threadIdx.x < st) red[threadIdx.x] = fmaxf(red[threadIdx.x], red[threadIdx.x + st]);
        __syncthreads();
    }
    m = red[0]; __syncthreads();
    float s = 0.f;
    for (int n = threadIdx.x; n < NPIX; n += 256) s += __expf(logit[n] - m);
    red[threadIdx.x] = s; __syncthreads();
    for (int st = 128; st > 0; st >>= 1) {
        if (threadIdx.x < st) red[threadIdx.x] += red[threadIdx.x + st];
        __syncthreads();
    }
    float inv = 1.f / red[0];
    float* po = Pi + (long)(b * NHEADS + h) * NPIX;
    for (int n = threadIdx.x; n < NPIX; n += 256) po[n] = __expf(logit[n] - m) * inv;
}

__global__ void tssa_dots_kernel(const float* __restrict__ qkv, const float* __restrict__ Pi,
                                 float* __restrict__ attn) {
    __shared__ float sm[256];
    int b = blockIdx.x >> 2, h = blockIdx.x & 3;
    const float* kb = qkv + (long)b * 768 * NPIX + (long)(CDIM + h * HD) * NPIX;
    const float* pb = Pi + (long)(b * NHEADS + h) * NPIX;
    int d = threadIdx.x & 63, chunk = threadIdx.x >> 6;
    float s = 0.f;
    for (int n = chunk * 256; n < chunk * 256 + 256; ++n) {
        float kv = kb[(long)d * NPIX + n];
        s += pb[n] * kv * kv;
    }
    sm[threadIdx.x] = s; __syncthreads();
    if (threadIdx.x < 64) {
        float t = sm[threadIdx.x] + sm[threadIdx.x + 64] + sm[threadIdx.x + 128] + sm[threadIdx.x + 192];
        attn[(long)(b * NHEADS + h) * HD + threadIdx.x] = 1.f / (1.f + t);
    }
}

__global__ void tssa_out_kernel(const float* __restrict__ qkv, const float* __restrict__ Pi,
                                const float* __restrict__ attn, float* __restrict__ feats, int si) {
    long idx = (long)blockIdx.x * 256 + threadIdx.x;
    int n = (int)(idx & (NPIX - 1));
    long bc = idx >> 10;
    int c = (int)(bc & 255), b = (int)(bc >> 8);
    int h = c >> 6, d = c & 63;
    float v  = qkv[(long)b * 768 * NPIX + (long)(512 + c) * NPIX + n];
    float pi = Pi[(long)(b * NHEADS + h) * NPIX + n];
    float at = attn[(long)(b * NHEADS + h) * HD + d];
    feats[(long)b * CDIM * LSEQ + (long)c * LSEQ + (long)si * NPIX + n] = -(v * pi) * at;
}

__global__ void mean_scales_kernel(const float* __restrict__ fused, float* __restrict__ avg) {
    long idx = (long)blockIdx.x * 256 + threadIdx.x;
    int n = (int)(idx & (NPIX - 1));
    long bc = idx >> 10;
    const float* p = fused + bc * LSEQ;
    avg[idx] = (p[n] + p[NPIX + n] + p[2 * NPIX + n]) * (1.f / 3.f);
}

__global__ void add_kernel(const float* __restrict__ a, const float* __restrict__ b2,
                           float* __restrict__ y, long n) {
    long i = (long)blockIdx.x * 256 + threadIdx.x;
    if (i < n) y[i] = a[i] + b2[i];
}

__global__ void mona_dw_kernel(const float* __restrict__ x,
                               const float* __restrict__ w3, const float* __restrict__ b3,
                               const float* __restrict__ w5, const float* __restrict__ b5,
                               const float* __restrict__ w7, const float* __restrict__ b7,
                               float* __restrict__ y, int C) {
    long idx = (long)blockIdx.x * 256 + threadIdx.x;
    int j = (int)(idx & 31), i = (int)((idx >> 5) & 31);
    long bc = idx >> 10; int c = (int)(bc % C);
    const float* p = x + bc * NPIX;
    float a3 = b3[c], a5 = b5[c], a7 = b7[c];
    for (int di = -3; di <= 3; ++di) {
        int ii = i + di; if (ii < 0 || ii >= 32) continue;
        for (int dj = -3; dj <= 3; ++dj) {
            int jj = j + dj; if (jj < 0 || jj >= 32) continue;
            float v = p[ii * 32 + jj];
            a7 += v * w7[(long)c * 49 + (di + 3) * 7 + (dj + 3)];
            if (di >= -2 && di <= 2 && dj >= -2 && dj <= 2)
                a5 += v * w5[(long)c * 25 + (di + 2) * 5 + (dj + 2)];
            if (di >= -1 && di <= 1 && dj >= -1 && dj <= 1)
                a3 += v * w3[(long)c * 9 + (di + 1) * 3 + (dj + 1)];
        }
    }
    y[idx] = (a3 + a5 + a7) * (1.f / 3.f) + p[i * 32 + j];
}

__global__ void dw3_kernel(const float* __restrict__ x, const float* __restrict__ w,
                           float* __restrict__ y, int C) {
    long idx = (long)blockIdx.x * 256 + threadIdx.x;
    int j = (int)(idx & 31), i = (int)((idx >> 5) & 31);
    long bc = idx >> 10; int c = (int)(bc % C);
    const float* p = x + bc * NPIX;
    float a = 0.f;
    #pragma unroll
    for (int di = -1; di <= 1; ++di) {
        int ii = i + di; if (ii < 0 || ii >= 32) continue;
        #pragma unroll
        for (int dj = -1; dj <= 1; ++dj) {
            int jj = j + dj; if (jj < 0 || jj >= 32) continue;
            a += p[ii * 32 + jj] * w[(long)c * 9 + (di + 1) * 3 + (dj + 1)];
        }
    }
    y[idx] = a;
}

__global__ void layernorm_c_kernel(float* __restrict__ u, const float* __restrict__ w,
                                   const float* __restrict__ bln) {
    long idx = (long)blockIdx.x * 256 + threadIdx.x;
    if (idx >= (long)BDIM * NPIX) return;
    int b = (int)(idx >> 10), n = (int)(idx & (NPIX - 1));
    float* p = u + (long)b * CDIM * NPIX + n;
    float s = 0.f, ss = 0.f;
    for (int c = 0; c < CDIM; ++c) { float v = p[(long)c * NPIX]; s += v; ss += v * v; }
    float mu = s / CDIM;
    float var = ss / CDIM - mu * mu;
    float inv = rsqrtf(var + 1e-5f);
    for (int c = 0; c < CDIM; ++c) {
        float v = p[(long)c * NPIX];
        p[(long)c * NPIX] = (v - mu) * inv * w[c] + bln[c];
    }
}

__global__ void hmona_combine_kernel(const float* __restrict__ xr,
                                     const float* __restrict__ u0, const float* __restrict__ u1,
                                     const float* __restrict__ u2, const float* __restrict__ fw,
                                     const float* __restrict__ gamma, float* __restrict__ dst) {
    long idx = (long)blockIdx.x * 256 + threadIdx.x;
    long bc = idx >> 10; int c = (int)(bc & 255);
    float ws_ = fw[0] * u0[idx] + fw[1] * u1[idx] + fw[2] * u2[idx];
    dst[idx] = xr[idx] + ws_ * gamma[c];
}

__global__ void gelu_mul_kernel(const float* __restrict__ dwv, float* __restrict__ gp) {
    long idx = (long)blockIdx.x * 256 + threadIdx.x;   // B*512*N
    int n = (int)(idx & (NPIX - 1));
    long bc = idx >> 10;
    int c = (int)(bc & 511), b = (int)(bc >> 9);
    const float* base = dwv + (long)b * 1024 * NPIX;
    float x1 = base[(long)c * NPIX + n];
    float x2 = base[(long)(512 + c) * NPIX + n];
    float g = 0.5f * x1 * (1.f + erff(x1 * 0.70710678f));
    gp[idx] = g * x2;
}

__global__ void gate_apply_kernel(float* __restrict__ x, const float* __restrict__ f,
                                  const float* __restrict__ g) {
    long idx = (long)blockIdx.x * 256 + threadIdx.x;
    long bc = idx >> 10;
    x[idx] += f[idx] * g[bc];
}

// ---------------- per-patch 8x8 rFFT2 * W -> irFFT2 ----------------
__constant__ float COS8[8] = {1.f, 0.70710678f, 0.f, -0.70710678f, -1.f, -0.70710678f, 0.f, 0.70710678f};
__constant__ float SIN8[8] = {0.f, 0.70710678f, 1.f, 0.70710678f, 0.f, -0.70710678f, -1.f, -0.70710678f};

__global__ __launch_bounds__(256)
void fft_patch_kernel(const float* __restrict__ xin, const float* __restrict__ fw,
                      float* __restrict__ xout) {
    __shared__ float re[32][8][5];
    __shared__ float im[32][8][5];
    int lp = threadIdx.x >> 3;    // local patch
    int r  = threadIdx.x & 7;     // row within patch
    long pid = (long)blockIdx.x * 32 + lp;      // B*C*16 patches
    int pw = (int)(pid & 3), ph = (int)((pid >> 2) & 3);
    long bc = pid >> 4; int c = (int)(bc & 255);
    const float* p = xin + bc * NPIX + (ph * 8 + r) * 32 + pw * 8;
    float v[8];
    #pragma unroll
    for (int j = 0; j < 8; ++j) v[j] = p[j];
    #pragma unroll
    for (int k = 0; k < 5; ++k) {              // row rfft
        float sr = 0.f, si = 0.f;
        #pragma unroll
        for (int j = 0; j < 8; ++j) {
            int t = (k * j) & 7;
            sr += v[j] * COS8[t];
            si -= v[j] * SIN8[t];
        }
        re[lp][r][k] = sr; im[lp][r][k] = si;
    }
    __syncthreads();
    float zr[5], zi[5];
    #pragma unroll
    for (int k = 0; k < 5; ++k) {              // column fft (freq t = r) * weight
        float sr = 0.f, si = 0.f;
        #pragma unroll
        for (int j = 0; j < 8; ++j) {
            int t = (r * j) & 7;
            float cr = COS8[t], ci = -SIN8[t];
            float ar = re[lp][j][k], ai = im[lp][j][k];
            sr += ar * cr - ai * ci;
            si += ar * ci + ai * cr;
        }
        float wgt = fw[(long)c * 40 + r * 5 + k];
        zr[k] = sr * wgt; zi[k] = si * wgt;
    }
    __syncthreads();
    #pragma unroll
    for (int k = 0; k < 5; ++k) { re[lp][r][k] = zr[k]; im[lp][r][k] = zi[k]; }
    __syncthreads();
    #pragma unroll
    for (int k = 0; k < 5; ++k) {              // inverse column fft (spatial row j = r)
        float sr = 0.f, si = 0.f;
        #pragma unroll
        for (int t = 0; t < 8; ++t) {
            int a = (r * t) & 7;
            float cr = COS8[a], ci = SIN8[a];
            float ar = re[lp][t][k], ai = im[lp][t][k];
            sr += ar * cr - ai * ci;
            si += ar * ci + ai * cr;
        }
        zr[k] = sr * 0.125f; zi[k] = si * 0.125f;
    }
    float* q = xout + bc * NPIX + (ph * 8 + r) * 32 + pw * 8;
    #pragma unroll
    for (int w = 0; w < 8; ++w) {              // irfft row
        float s = zr[0];
        #pragma unroll
        for (int k = 1; k < 4; ++k) {
            int a = (k * w) & 7;
            s += 2.f * (zr[k] * COS8[a] - zi[k] * SIN8[a]);
        }
        int a4 = (4 * w) & 7;
        s += zr[4] * COS8[a4] - zi[4] * SIN8[a4];
        q[w] = s * 0.125f;
    }
}

// =====================================================================
// host orchestration
// =====================================================================
extern "C" void kernel_launch(void* const* d_in, const int* in_sizes, int n_in,
                              void* d_out, int out_size, void* d_ws, size_t ws_size,
                              hipStream_t stream) {
    (void)in_sizes; (void)n_in; (void)out_size; (void)ws_size;
    auto P = [&](int i) -> const float* { return (const float*)d_in[i]; };
    float* ws = (float*)d_ws;

    // ---- workspace layout (float offsets), total ~19.2M floats (~76.6 MB) ----
    const long o_mean = 0;
    const long o_g1   = o_mean + 1024;
    const long o_imp  = o_g1   + 256;
    const long o_pi   = o_imp  + 64;
    const long o_attn = o_pi   + 16384;
    const long o_gate = o_attn + 1024;
    const long o_X1   = o_gate + 1024;
    const long o_pool = o_X1   + 1048576;
    const long o_xsup = o_pool + 262144;     // also mean-over-scales
    const long o_qkvs = o_xsup + 1048576;    // also flash O
    const long o_feat = o_qkvs + 3145728;    // also fused / x2
    const long o_big  = o_feat + 3145728;    // qkv_m / hmona & ffn scratch
    const long o_xr   = o_big  + 9437184;

    auto gemm = [&](const float* Wm, const float* X0, const float* X1g, int ksplit,
                    float* Y, const float* bias, const float* addsrc,
                    int M, int N, int K, long sx0, long sx1, long sy, long sadd) {
        dim3 g(N / 128, M / 32, BDIM);
        gemm_wmma_kernel<<<g, dim3(256), 0, stream>>>(Wm, X0, X1g, ksplit, Y, bias,
                                                      addsrc, M, N, K, sx0, sx1, sy, sadd);
    };

    const float* x = P(126);   // input tensor (B,C,32,32)
    const long ECNT = (long)BDIM * CDIM * NPIX;     // 1,048,576
    const int EB = (int)(ECNT / 256);

    // ---- adaptive_dyt #1 ----
    mean_hw_kernel<<<BDIM * CDIM, 256, 0, stream>>>(x, ws + o_mean);
    gemv_kernel<<<BDIM, 64, 0, stream>>>(P(11), P(10), ws + o_mean, ws + o_g1, 64, 256, 1);
    gemv_kernel<<<BDIM, 32, 0, stream>>>(P(13), P(12), ws + o_g1, ws + o_imp, 3, 64, 0);
    softmax3_kernel<<<1, 32, 0, stream>>>(ws + o_imp);
    dyt_apply_kernel<<<EB, 256, 0, stream>>>(x, P(8), ws + o_imp, P(14), P(9), ws + o_X1);

    // ---- TSSA multi-scale features -> feats (B,256,3N) ----
    for (int i = 0; i < 3; ++i) {
        int s = (i == 0) ? 1 : (i == 1) ? 2 : 4;
        const float* xs = ws + o_X1;
        if (s > 1) {
            long tot = (long)BDIM * CDIM * (32 / s) * (32 / s);
            avgpool_kernel<<<(int)((tot + 255) / 256), 256, 0, stream>>>(ws + o_X1, ws + o_pool, s);
            upsample_kernel<<<EB, 256, 0, stream>>>(ws + o_pool, ws + o_xsup, s);
            xs = ws + o_xsup;
        }
        gemm(P(4) + (long)i * 768 * 256, xs, nullptr, 1 << 30, ws + o_qkvs, nullptr, nullptr,
             768, NPIX, 256, (long)CDIM * NPIX, 0, (long)768 * NPIX, 0);
        tssa_pi_kernel<<<BDIM * NHEADS, 256, 0, stream>>>(ws + o_qkvs, P(5), i, ws + o_pi);
        tssa_dots_kernel<<<BDIM * NHEADS, 256, 0, stream>>>(ws + o_qkvs, ws + o_pi, ws + o_attn);
        tssa_out_kernel<<<EB, 256, 0, stream>>>(ws + o_qkvs, ws + o_pi, ws + o_attn, ws + o_feat, i);
    }

    // ---- MHA fuse: qkv proj -> flash -> out proj -> mean scales -> to_out ----
    gemm(P(1), ws + o_feat, nullptr, 1 << 30, ws + o_big, P(0), nullptr,
         768, LSEQ, 256, (long)CDIM * LSEQ, 0, (long)768 * LSEQ, 0);
    flash_attn_kernel<<<dim3(LSEQ / 128, NHEADS, BDIM), 256, 0, stream>>>(ws + o_big, ws + o_qkvs, LSEQ, 0.125f);
    gemm(P(3), ws + o_qkvs, nullptr, 1 << 30, ws + o_feat, P(2), nullptr,
         256, LSEQ, 256, (long)CDIM * LSEQ, 0, (long)CDIM * LSEQ, 0);
    mean_scales_kernel<<<EB, 256, 0, stream>>>(ws + o_feat, ws + o_xsup);
    gemm(P(7), ws + o_xsup, nullptr, 1 << 30, ws + o_X1, P(6), nullptr,
         256, NPIX, 256, (long)CDIM * NPIX, 0, (long)CDIM * NPIX, 0);
    add_kernel<<<EB, 256, 0, stream>>>(x, ws + o_X1, ws + o_xr, ECNT);

    // ---- HMONA (shared for mona1 / mona2) ----
    auto run_hmona = [&](int base, float* dst) {
        float* xr = ws + o_xr;
        float* s_d  = ws + o_big;
        float* s_mt = ws + o_big + 1048576;
        float* s_u[3] = { ws + o_big + 2 * 1048576, ws + o_big + 3 * 1048576, ws + o_big + 4 * 1048576 };
        float* s_curA = ws + o_big + 5 * 1048576;
        float* s_curB = ws + o_big + 6 * 1048576;
        const float* cur = xr;
        int lvbase = base + 6;
        for (int lv = 0; lv < 3; ++lv) {
            int nd = (lv == 0) ? 256 : (lv == 1) ? 128 : 64;
            int li = lvbase + lv * 14;
            gemm(P(li + 1), cur, nullptr, 1 << 30, s_d, P(li + 0), nullptr,
                 nd, NPIX, 256, (long)CDIM * NPIX, 0, (long)nd * NPIX, 0);
            long tot = (long)BDIM * nd * NPIX;
            mona_dw_kernel<<<(int)(tot / 256), 256, 0, stream>>>(s_d, P(li + 5), P(li + 4),
                P(li + 7), P(li + 6), P(li + 9), P(li + 8), s_mt, nd);
            gemm(P(li + 11), s_mt, nullptr, 1 << 30, s_d, P(li + 10), s_mt,
                 nd, NPIX, nd, (long)nd * NPIX, 0, (long)nd * NPIX, (long)nd * NPIX);
            gemm(P(li + 13), s_d, nullptr, 1 << 30, s_u[lv], P(li + 12), nullptr,
                 CDIM, NPIX, nd, (long)nd * NPIX, 0, (long)CDIM * NPIX, 0);
            layernorm_c_kernel<<<(BDIM * NPIX) / 256, 256, 0, stream>>>(s_u[lv], P(li + 3), P(li + 2));
            if (lv < 2) {
                float* nxt = (lv == 0) ? s_curA : s_curB;
                gemm(P(base + 3 + lv), cur, s_u[lv], 256, nxt, P(base + 1 + lv), nullptr,
                     CDIM, NPIX, 512, (long)CDIM * NPIX, (long)CDIM * NPIX, (long)CDIM * NPIX, 0);
                cur = nxt;
            }
        }
        hmona_combine_kernel<<<EB, 256, 0, stream>>>(xr, s_u[0], s_u[1], s_u[2],
                                                     P(base), P(base + 5), dst);
    };
    run_hmona(30, ws + o_xr);   // mona1 (in-place on xr)

    // ---- adaptive_dyt #2 ----
    mean_hw_kernel<<<BDIM * CDIM, 256, 0, stream>>>(ws + o_xr, ws + o_mean);
    gemv_kernel<<<BDIM, 64, 0, stream>>>(P(18), P(17), ws + o_mean, ws + o_g1, 64, 256, 1);
    gemv_kernel<<<BDIM, 32, 0, stream>>>(P(20), P(19), ws + o_g1, ws + o_imp, 3, 64, 0);
    softmax3_kernel<<<1, 32, 0, stream>>>(ws + o_imp);
    float* x2buf = ws + o_feat;
    dyt_apply_kernel<<<EB, 256, 0, stream>>>(ws + o_xr, P(15), ws + o_imp, P(21), P(16), x2buf);

    // ---- EDFFN ----
    float* e_mid = ws + o_big;
    float* e_dw  = ws + o_big + 4194304;
    float* e_gp  = e_mid;
    float* e_f   = e_mid + 2097152;
    float* e_f2  = e_dw;
    gemm(P(24), x2buf, nullptr, 1 << 30, e_mid, nullptr, nullptr,
         1024, NPIX, 256, (long)CDIM * NPIX, 0, (long)1024 * NPIX, 0);
    dw3_kernel<<<(int)((long)BDIM * 1024 * NPIX / 256), 256, 0, stream>>>(e_mid, P(22), e_dw, 1024);
    gelu_mul_kernel<<<(int)((long)BDIM * 512 * NPIX / 256), 256, 0, stream>>>(e_dw, e_gp);
    gemm(P(25), e_gp, nullptr, 1 << 30, e_f, nullptr, nullptr,
         256, NPIX, 512, (long)512 * NPIX, 0, (long)CDIM * NPIX, 0);
    fft_patch_kernel<<<BDIM * CDIM * 16 / 32, 256, 0, stream>>>(e_f, P(23), e_f2);

    // ---- channel gate + residual ----
    mean_hw_kernel<<<BDIM * CDIM, 256, 0, stream>>>(ws + o_xr, ws + o_mean);
    gemv_kernel<<<BDIM, 64, 0, stream>>>(P(27), P(26), ws + o_mean, ws + o_g1, 64, 256, 1);
    gemv_kernel<<<BDIM, 256, 0, stream>>>(P(29), P(28), ws + o_g1, ws + o_gate, 256, 64, 2);
    gate_apply_kernel<<<EB, 256, 0, stream>>>(ws + o_xr, e_f2, ws + o_gate);

    // ---- HMONA #2, final output straight into d_out ----
    run_hmona(78, (float*)d_out);
}
